// DecoderRNN_77988016161281
// MI455X (gfx1250) — compile-verified, matrix operand order checked by reading, not run on hardware
//
#include <hip/hip_runtime.h>
#include <stdint.h>

#define kB 512
#define kT 256
#define kH 512
#define kV 7
#define k4H 2048

typedef __attribute__((ext_vector_type(16))) __bf16 v16bf;
typedef __attribute__((ext_vector_type(8)))  float  v8f;
typedef int v4i_vec __attribute__((vector_size(16)));   // async builtin's pointee type

// ---------- small device helpers ----------

__device__ __forceinline__ uint16_t f2bf(float x) {
  union { float f; uint32_t u; } c; c.f = x;
  uint32_t r = c.u + 0x7FFFu + ((c.u >> 16) & 1u);   // round-to-nearest-even
  return (uint16_t)(r >> 16);
}

__device__ __forceinline__ float sigmoidf_(float x) {
  return 1.0f / (1.0f + __expf(-x));
}

// Load a 16-element bf16 WMMA fragment: two 16B chunks at +0 and +16 elements.
// Per-lane A/B layout (16-bit, 16x32): lane group hi = lane>>4 selects
// k-base hi*8; halves 0..7 -> k = hi*8+e, halves 8..15 -> k = hi*8+16+(e-8).
__device__ __forceinline__ v16bf load_frag16(const uint16_t* p) {
  union { uint4 q[2]; v16bf v; } u;
  u.q[0] = *(const uint4*)(p);
  u.q[1] = *(const uint4*)(p + 16);
  return u.v;
}

// Async global->LDS 16B copy (CDNA5 GLOBAL_LOAD_ASYNC_TO_LDS_B128), with
// portable fallback if the builtin is absent on this toolchain.
__device__ __forceinline__ void stage16(const uint16_t* g, uint16_t* l) {
#if __has_builtin(__builtin_amdgcn_global_load_async_to_lds_b128)
  __builtin_amdgcn_global_load_async_to_lds_b128(
      (__attribute__((address_space(1))) v4i_vec*)(g),
      (__attribute__((address_space(3))) v4i_vec*)(l), 0, 0);
#else
  *(uint4*)l = *(const uint4*)g;
#endif
}

__device__ __forceinline__ void wait_async0() {
#if __has_builtin(__builtin_amdgcn_global_load_async_to_lds_b128)
#if __has_builtin(__builtin_amdgcn_s_wait_asynccnt)
  __builtin_amdgcn_s_wait_asynccnt(0);
#else
  asm volatile("s_wait_asynccnt 0" ::: "memory");
#endif
#endif
}

// ---------- one-time setup kernels (re-run each call: deterministic) ----------

__global__ __launch_bounds__(256) void init_kernel(
    const float* __restrict__ h0, const float* __restrict__ c0,
    const float* __restrict__ onehot,   // [B,T,V]
    const float* __restrict__ b_ih, const float* __restrict__ b_hh,
    float* __restrict__ h_f32, float* __restrict__ c_f32,
    uint16_t* __restrict__ h_bf0, float* __restrict__ bias,
    int* __restrict__ idx) {
  int i = blockIdx.x * 256 + threadIdx.x;
  if (i < kB * kH) {
    float hv = h0[i];
    h_f32[i] = hv;
    c_f32[i] = c0[i];
    h_bf0[i] = f2bf(hv);
  }
  if (i < k4H) bias[i] = b_ih[i] + b_hh[i];
  if (i < kB) {
    const float* row = onehot + (size_t)i * kT * kV;  // t = 0
    int bi = 0; float best = row[0];
#pragma unroll
    for (int v = 1; v < kV; ++v) { float x = row[v]; if (x > best) { best = x; bi = v; } }
    idx[i] = bi;
  }
}

__global__ __launch_bounds__(256) void convert_whh_kernel(
    const float* __restrict__ W_hh, uint16_t* __restrict__ Whh_bf) {
  int i = blockIdx.x * 256 + threadIdx.x;
  if (i < k4H * kH) Whh_bf[i] = f2bf(W_hh[i]);
}

// ---------- per-step fused GEMM + LSTM cell ----------
// grid = (B/64, H/32), block = 256 (8 waves).
// Wave w: m_tile = (w&3)*16, n_tile = (w>>2)*16. Each wave accumulates the
// same 16x16 (m,n) tile for all 4 gates (n, n+H, n+2H, n+3H) so the cell
// update is register-local. The full 64-row x 512-K h tile (64 KB bf16,
// contiguous in global) is staged to LDS ONCE with the async copy engine;
// the K loop then runs barrier-free: ds_load + global_load + wmma only.

__global__ __launch_bounds__(256) void lstm_step_kernel(
    const uint16_t* __restrict__ h_in_bf,   // [B,H] bf16 (double-buffered)
    uint16_t* __restrict__ h_out_bf,        // [B,H] bf16
    float* __restrict__ h_f32,              // [B,H]
    float* __restrict__ c_f32,              // [B,H]
    const uint16_t* __restrict__ Whh_bf,    // [4H,H] bf16 row-major
    const float* __restrict__ bias,         // [4H] (= b_ih + b_hh)
    const float* __restrict__ W_ih,         // [4H,V] fp32 row-major
    const int* __restrict__ idx) {          // [B] previous argmax
  __shared__ __align__(16) uint16_t ldsA[64 * kH];  // 64 batch rows x 512 K (64 KB)

  const int tid  = threadIdx.x;
  const int wave = tid >> 5;
  const int lane = tid & 31;
  const int mf   = lane & 15;
  const int hi   = lane >> 4;

  const int m_blk  = blockIdx.x * 64;
  const int n_blk  = blockIdx.y * 32;
  const int m_tile = (wave & 3) * 16;
  const int n_tile = (wave >> 2) * 16;
  const int n_g    = n_blk + n_tile + mf;           // B-fragment column -> W_hh row (mod H)

  // Stage the whole 64x512 h tile (contiguous 64 KB): 16 x b128 per thread.
  {
    const uint16_t* gbase = h_in_bf + (size_t)m_blk * kH;
#pragma unroll
    for (int s = 0; s < 16; ++s) {
      const int c = tid + s * 256;                  // 16B-chunk id, 0..4095
      stage16(gbase + c * 8, &ldsA[c * 8]);
    }
  }

  v8f acc_i = {}, acc_f = {}, acc_g = {}, acc_o = {};

  const uint16_t* wrow0 = Whh_bf + (size_t)(0 * kH + n_g) * kH;
  const uint16_t* wrow1 = Whh_bf + (size_t)(1 * kH + n_g) * kH;
  const uint16_t* wrow2 = Whh_bf + (size_t)(2 * kH + n_g) * kH;
  const uint16_t* wrow3 = Whh_bf + (size_t)(3 * kH + n_g) * kH;

  wait_async0();      // own wave's async transfers done
  __syncthreads();    // all waves' transfers done (each waited on its own)

  const uint16_t* arow = &ldsA[(m_tile + mf) * kH + hi * 8];

#pragma unroll
  for (int k0 = 0; k0 < kH; k0 += 32) {
    v16bf a = load_frag16(arow + k0);

    if (k0 + 32 < kH) __builtin_prefetch(wrow0 + k0 + 32, 0, 1);  // global_prefetch

    v16bf b0 = load_frag16(wrow0 + k0 + hi * 8);
    acc_i = __builtin_amdgcn_wmma_f32_16x16x32_bf16(false, a, false, b0,
                                                    (short)0, acc_i, false, false);
    v16bf b1 = load_frag16(wrow1 + k0 + hi * 8);
    acc_f = __builtin_amdgcn_wmma_f32_16x16x32_bf16(false, a, false, b1,
                                                    (short)0, acc_f, false, false);
    v16bf b2 = load_frag16(wrow2 + k0 + hi * 8);
    acc_g = __builtin_amdgcn_wmma_f32_16x16x32_bf16(false, a, false, b2,
                                                    (short)0, acc_g, false, false);
    v16bf b3 = load_frag16(wrow3 + k0 + hi * 8);
    acc_o = __builtin_amdgcn_wmma_f32_16x16x32_bf16(false, a, false, b3,
                                                    (short)0, acc_o, false, false);
  }

  // Epilogue: C/D layout -> element r: m = r + hi*8 (within tile), n = mf.
  const int n = n_blk + n_tile + mf;
  const float bi_i = bias[0 * kH + n];
  const float bi_f = bias[1 * kH + n];
  const float bi_g = bias[2 * kH + n];
  const float bi_o = bias[3 * kH + n];
#pragma unroll
  for (int r = 0; r < 8; ++r) {
    const int m = m_blk + m_tile + r + hi * 8;
    const int id = idx[m];
    const float gi = acc_i[r] + bi_i + W_ih[(size_t)(0 * kH + n) * kV + id];
    const float gf = acc_f[r] + bi_f + W_ih[(size_t)(1 * kH + n) * kV + id];
    const float gg = acc_g[r] + bi_g + W_ih[(size_t)(2 * kH + n) * kV + id];
    const float go = acc_o[r] + bi_o + W_ih[(size_t)(3 * kH + n) * kV + id];

    const size_t e = (size_t)m * kH + n;
    const float c_new = sigmoidf_(gf) * c_f32[e] + sigmoidf_(gi) * tanhf(gg);
    const float h_new = sigmoidf_(go) * tanhf(c_new);
    c_f32[e] = c_new;
    h_f32[e] = h_new;
    h_out_bf[e] = f2bf(h_new);
  }
}

// ---------- per-step output projection + argmax ----------
// One wave per batch row; 7 length-512 dots with shuffle reduction.

__global__ __launch_bounds__(256) void out_proj_kernel(
    const float* __restrict__ h_f32, const float* __restrict__ W_out,
    const float* __restrict__ b_out, float* __restrict__ out_logits,
    int* __restrict__ idx, int t) {
  const int wave = threadIdx.x >> 5;
  const int lane = threadIdx.x & 31;
  const int b = blockIdx.x * 8 + wave;

  float acc[kV];
#pragma unroll
  for (int v = 0; v < kV; ++v) acc[v] = 0.0f;

  const float* hrow = h_f32 + (size_t)b * kH;
  for (int k = lane; k < kH; k += 32) {
    const float hv = hrow[k];
#pragma unroll
    for (int v = 0; v < kV; ++v) acc[v] += hv * W_out[v * kH + k];
  }
#pragma unroll
  for (int v = 0; v < kV; ++v)
    for (int off = 16; off > 0; off >>= 1) acc[v] += __shfl_xor(acc[v], off, 32);

  if (lane == 0) {
    float* row = out_logits + (size_t)b * kT * kV + (size_t)t * kV;
    float best = -3.4e38f; int bi = 0;
#pragma unroll
    for (int v = 0; v < kV; ++v) {
      const float o = acc[v] + b_out[v];
      row[v] = o;
      if (o > best) { best = o; bi = v; }
    }
    idx[b] = bi;
  }
}

// ---------- finalize: in-place log_softmax + hT/cT copy ----------

__global__ __launch_bounds__(256) void finalize_kernel(
    const float* __restrict__ h_f32, const float* __restrict__ c_f32,
    float* __restrict__ out) {
  int i = blockIdx.x * 256 + threadIdx.x;
  if (i < kB * kT) {
    float* row = out + (size_t)i * kV;
    float m = row[0];
#pragma unroll
    for (int v = 1; v < kV; ++v) m = fmaxf(m, row[v]);
    float s = 0.0f;
#pragma unroll
    for (int v = 0; v < kV; ++v) s += __expf(row[v] - m);
    const float l = __logf(s);
#pragma unroll
    for (int v = 0; v < kV; ++v) row[v] = row[v] - m - l;
  }
  if (i < kB * kH) {
    out[(size_t)kB * kT * kV + i] = h_f32[i];
    out[(size_t)kB * kT * kV + (size_t)kB * kH + i] = c_f32[i];
  }
}

// ---------- host launcher ----------

extern "C" void kernel_launch(void* const* d_in, const int* in_sizes, int n_in,
                              void* d_out, int out_size, void* d_ws, size_t ws_size,
                              hipStream_t stream) {
  (void)in_sizes; (void)n_in; (void)out_size; (void)ws_size;
  const float* h0     = (const float*)d_in[0];
  const float* c0     = (const float*)d_in[1];
  const float* onehot = (const float*)d_in[2];
  const float* W_ih   = (const float*)d_in[3];
  const float* W_hh   = (const float*)d_in[4];
  const float* b_ih   = (const float*)d_in[5];
  const float* b_hh   = (const float*)d_in[6];
  const float* W_out  = (const float*)d_in[7];
  const float* b_out  = (const float*)d_in[8];
  float* out = (float*)d_out;

  // Workspace carve-out (256B-aligned slices), ~5.3 MB total.
  char* ws = (char*)d_ws;
  auto carve = [&ws](size_t bytes) -> char* {
    char* p = ws;
    ws += (bytes + 255) & ~(size_t)255;
    return p;
  };
  uint16_t* Whh_bf = (uint16_t*)carve((size_t)k4H * kH * 2);
  uint16_t* hbf0   = (uint16_t*)carve((size_t)kB * kH * 2);
  uint16_t* hbf1   = (uint16_t*)carve((size_t)kB * kH * 2);
  float*    h_f32  = (float*)carve((size_t)kB * kH * 4);
  float*    c_f32  = (float*)carve((size_t)kB * kH * 4);
  float*    bias   = (float*)carve((size_t)k4H * 4);
  int*      idx    = (int*)carve((size_t)kB * 4);

  init_kernel<<<(kB * kH + 255) / 256, 256, 0, stream>>>(
      h0, c0, onehot, b_ih, b_hh, h_f32, c_f32, hbf0, bias, idx);
  convert_whh_kernel<<<(k4H * kH + 255) / 256, 256, 0, stream>>>(W_hh, Whh_bf);

  for (int t = 0; t < kT; ++t) {
    const uint16_t* hin = (t & 1) ? hbf1 : hbf0;
    uint16_t*       hout = (t & 1) ? hbf0 : hbf1;
    lstm_step_kernel<<<dim3(kB / 64, kH / 32), 256, 0, stream>>>(
        hin, hout, h_f32, c_f32, Whh_bf, bias, W_ih, idx);
    out_proj_kernel<<<kB / 8, 256, 0, stream>>>(h_f32, W_out, b_out, out, idx, t);
  }

  finalize_kernel<<<(kB * kH + 255) / 256, 256, 0, stream>>>(h_f32, c_f32, out);
}
